// MLAAttention_69389491634335
// MI455X (gfx1250) — compile-verified, matrix operand order checked by reading, not run on hardware
//
#include <hip/hip_runtime.h>

// ---------------------------------------------------------------------------
// MLA forward for MI455X (gfx1250, wave32, WMMA 16x16x32 bf16,
// async global->LDS tile staging)
// ---------------------------------------------------------------------------

typedef __attribute__((ext_vector_type(16))) __bf16        v16bf;
typedef __attribute__((ext_vector_type(8)))  float         v8f;
typedef __attribute__((ext_vector_type(4)))  unsigned int  u32x4;
typedef __attribute__((ext_vector_type(4)))  float         f32x4;
typedef __attribute__((ext_vector_type(4)))  unsigned short u16x4;

static constexpr int BB  = 2;
static constexpr int SS  = 2048;
static constexpr int HH  = 4096;
static constexpr int NHH = 32;
static constexpr int HDD = 128;
static constexpr int QR  = 1536;
static constexpr int KVR = 512;
static constexpr int MS  = BB * SS;    // 4096 token rows
static constexpr int DM  = NHH * HDD;  // 4096

__device__ __forceinline__ unsigned short f2bf(float f) {
  unsigned int u = __float_as_uint(f);
  u += 0x7FFFu + ((u >> 16) & 1u);      // round to nearest even
  return (unsigned short)(u >> 16);
}

// async copy of 16B per lane from global to LDS (bypasses VGPRs, ASYNCcnt)
__device__ __forceinline__ void async_copy_b128(const void* gsrc, void* ldst) {
  unsigned lds_off = (unsigned)(size_t)ldst;   // low 32 bits of generic = LDS addr
  asm volatile("global_load_async_to_lds_b128 %0, %1, off"
               :: "v"(lds_off), "v"((unsigned long long)(size_t)gsrc)
               : "memory");
}
__device__ __forceinline__ void wait_async() {
  asm volatile("s_wait_asynccnt 0x0" ::: "memory");
}

union FragU { v16bf v; u32x4 q[2]; };

// ---------------------------------------------------------------------------
// fp32 -> bf16 conversion (4 elements / thread)
// ---------------------------------------------------------------------------
__global__ __launch_bounds__(256) void cvt_f32_bf16(const float* __restrict__ in,
                                                    unsigned short* __restrict__ out,
                                                    int n4) {
  int i = blockIdx.x * 256 + threadIdx.x;
  if (i >= n4) return;
  f32x4 f = ((const f32x4*)in)[i];
  u16x4 h;
  h[0] = f2bf(f[0]); h[1] = f2bf(f[1]); h[2] = f2bf(f[2]); h[3] = f2bf(f[3]);
  ((u16x4*)out)[i] = h;
}

// ---------------------------------------------------------------------------
// bf16 GEMM: C[M,N] = A[M,K] * B[K,N], f32 accumulate.
// Block tile 128x128, K-step 64 (two WMMA k-steps per barrier).
// 256 threads = 8 waves (4 x 2), each wave computes 32x64 (2x4 tiles).
// A tiles staged via async global->LDS copies into double-buffered LDS;
// B tiles staged via registers (transpose) into double-buffered LDS.
// OUTMODE: 0 = bf16 out, 1 = f32 out, 2 = both.
// ---------------------------------------------------------------------------
template <int OUTMODE>
__global__ __launch_bounds__(256) void gemm_bf16(const unsigned short* __restrict__ A,
                                                 const unsigned short* __restrict__ Bm,
                                                 unsigned short* __restrict__ outB,
                                                 float* __restrict__ outF,
                                                 int M, int N, int K) {
  __shared__ __align__(16) unsigned short ldsA[2][128 * 64];   // row-major [m][k]
  __shared__ __align__(16) unsigned short ldsBT[2][128 * 64];  // transposed [n][k]

  const int t    = threadIdx.x;
  const int lane = t & 31;
  const int wave = t >> 5;
  const int wm   = wave >> 1;   // 0..3
  const int wn   = wave & 1;    // 0..1
  const int bM   = blockIdx.y * 128;
  const int bN   = blockIdx.x * 128;
  const int hl   = lane >> 4;   // which 16-lane half
  const int l16  = lane & 15;
  const int b0   = hl * 8;      // K sub-base for this half

  v8f acc[2][4];
#pragma unroll
  for (int mi = 0; mi < 2; ++mi)
#pragma unroll
    for (int ni = 0; ni < 4; ++ni)
#pragma unroll
      for (int e = 0; e < 8; ++e) acc[mi][ni][e] = 0.0f;

  // A tile 128x64 = 1024 x 16B : async copy, 4 per thread
  auto asyncLoadA = [&](int kb, int buf) {
#pragma unroll
    for (int rep = 0; rep < 4; ++rep) {
      int idx = t + rep * 256;
      async_copy_b128(A + (size_t)(bM + (idx >> 3)) * K + kb * 64 + (idx & 7) * 8,
                      &ldsA[buf][(idx >> 3) * 64 + (idx & 7) * 8]);
    }
  };
  // B tile 64x128 = 1024 x 16B : register staged + transposed store
  u32x4 rb[4];
  auto loadB = [&](int kb) {
#pragma unroll
    for (int rep = 0; rep < 4; ++rep) {
      int idx = t + rep * 256;
      rb[rep] = *(const u32x4*)(Bm + (size_t)(kb * 64 + (idx >> 4)) * N + bN + (idx & 15) * 8);
    }
  };
  auto storeB = [&](int buf) {
#pragma unroll
    for (int rep = 0; rep < 4; ++rep) {
      int idx = t + rep * 256;
      int r = idx >> 4, c = idx & 15;
      const unsigned short* hp = (const unsigned short*)&rb[rep];
#pragma unroll
      for (int j = 0; j < 8; ++j)
        ldsBT[buf][(c * 8 + j) * 64 + r] = hp[j];
    }
  };

  const int nK = K >> 6;
  asyncLoadA(0, 0);
  loadB(0);
  storeB(0);
  wait_async();
  __syncthreads();

  for (int kb = 0; kb < nK; ++kb) {
    const int cur = kb & 1, nxt = cur ^ 1;
    if (kb + 1 < nK) {              // stage next tile while computing this one
      asyncLoadA(kb + 1, nxt);
      loadB(kb + 1);
    }
    if (kb + 2 < nK) {              // L2 prefetch two tiles ahead
      __builtin_prefetch(A + (size_t)(bM + (t >> 3)) * K + (kb + 2) * 64 + (t & 7) * 8, 0, 3);
      __builtin_prefetch(Bm + (size_t)((kb + 2) * 64 + (t >> 4)) * N + bN + (t & 15) * 8, 0, 3);
    }
#pragma unroll
    for (int s = 0; s < 2; ++s) {
      FragU afr[2], bfr[4];
#pragma unroll
      for (int mi = 0; mi < 2; ++mi) {
        int row = wm * 32 + mi * 16 + l16;
        afr[mi].q[0] = *(const u32x4*)&ldsA[cur][row * 64 + s * 32 + b0];
        afr[mi].q[1] = *(const u32x4*)&ldsA[cur][row * 64 + s * 32 + 16 + b0];
      }
#pragma unroll
      for (int ni = 0; ni < 4; ++ni) {
        int col = wn * 64 + ni * 16 + l16;
        bfr[ni].q[0] = *(const u32x4*)&ldsBT[cur][col * 64 + s * 32 + b0];
        bfr[ni].q[1] = *(const u32x4*)&ldsBT[cur][col * 64 + s * 32 + 16 + b0];
      }
#pragma unroll
      for (int mi = 0; mi < 2; ++mi)
#pragma unroll
        for (int ni = 0; ni < 4; ++ni)
          acc[mi][ni] = __builtin_amdgcn_wmma_f32_16x16x32_bf16(
              false, afr[mi].v, false, bfr[ni].v, (short)0, acc[mi][ni], false, false);
    }
    if (kb + 1 < nK) storeB(nxt);   // nxt readers all left it at previous barrier
    wait_async();                    // next A tile resident in LDS
    __syncthreads();                 // all waves done with cur + nxt fully staged
  }

  // Epilogue (branch-free): lanes 0-15 of a fixed e cover 16 consecutive cols
#pragma unroll
  for (int mi = 0; mi < 2; ++mi)
#pragma unroll
    for (int ni = 0; ni < 4; ++ni)
#pragma unroll
      for (int e = 0; e < 8; ++e) {
        int grow = bM + wm * 32 + mi * 16 + e + hl * 8;
        int gcol = bN + wn * 64 + ni * 16 + l16;
        float val = acc[mi][ni][e];
        if constexpr (OUTMODE == 1 || OUTMODE == 2)
          outF[(size_t)grow * N + gcol] = val;
        if constexpr (OUTMODE == 0 || OUTMODE == 2)
          outB[(size_t)grow * N + gcol] = f2bf(val);
      }
}

// ---------------------------------------------------------------------------
// Flash attention, causal, one (b, h, 64-row Q block) per CTA.
// 128 threads = 4 waves, each wave owns 16 query rows.
// KV chunks of 32 keys; K async-staged row-major (double-buffered LDS),
// V register-staged transposed (double-buffered LDS); L2 prefetch chunk+2.
// ---------------------------------------------------------------------------
__global__ __launch_bounds__(128) void mla_attn(const unsigned short* __restrict__ Q,
                                                const unsigned short* __restrict__ Kt,
                                                const unsigned short* __restrict__ Vt,
                                                unsigned short* __restrict__ Ctx) {
  __shared__ __align__(16) unsigned short ldsK[2][32 * 128];   // [key][d]
  __shared__ __align__(16) unsigned short ldsVT[2][128 * 32];  // [d][key]
  __shared__ __align__(16) unsigned short ldsP[4 * 16 * 32];   // per-wave P tiles

  const int t    = threadIdx.x;
  const int lane = t & 31;
  const int wave = t >> 5;
  const int hl   = lane >> 4;
  const int l16  = lane & 15;
  const int b0   = hl * 8;
  const int qblk = blockIdx.x * 64;
  const int h    = blockIdx.y;
  const int b    = blockIdx.z;

  const size_t rowStride = (size_t)DM;                 // 4096 bf16 per token row
  const size_t base      = (size_t)b * SS * rowStride; // per-batch base
  const size_t headOff   = (size_t)h * HDD;

  // Q fragments for this wave's 16 rows (A layout, 4 K-chunks of 32 over HD=128)
  FragU qf[4];
  {
    int qr = qblk + wave * 16 + l16;
    const unsigned short* qrow = Q + base + (size_t)qr * rowStride + headOff;
#pragma unroll
    for (int c = 0; c < 4; ++c) {
      qf[c].q[0] = *(const u32x4*)(qrow + c * 32 + b0);
      qf[c].q[1] = *(const u32x4*)(qrow + c * 32 + 16 + b0);
    }
  }

  v8f O[8];
#pragma unroll
  for (int dt = 0; dt < 8; ++dt)
#pragma unroll
    for (int e = 0; e < 8; ++e) O[dt][e] = 0.0f;
  float mrow[8], lrow[8];
#pragma unroll
  for (int e = 0; e < 8; ++e) { mrow[e] = -1e30f; lrow[e] = 0.0f; }

  // K chunk: async copy (pure copy); V chunk: register staged + transpose
  auto asyncLoadK = [&](int kv0, int buf) {
#pragma unroll
    for (int rep = 0; rep < 4; ++rep) {
      int idx = t + rep * 128;
      int r = idx >> 4, c = idx & 15;
      async_copy_b128(Kt + base + (size_t)(kv0 + r) * rowStride + headOff + c * 8,
                      &ldsK[buf][r * 128 + c * 8]);
    }
  };
  u32x4 rv[4];
  auto loadV = [&](int kv0) {
#pragma unroll
    for (int rep = 0; rep < 4; ++rep) {
      int idx = t + rep * 128;
      int r = idx >> 4, c = idx & 15;
      rv[rep] = *(const u32x4*)(Vt + base + (size_t)(kv0 + r) * rowStride + headOff + c * 8);
    }
  };
  auto storeV = [&](int buf) {
#pragma unroll
    for (int rep = 0; rep < 4; ++rep) {
      int idx = t + rep * 128;
      int r = idx >> 4, c = idx & 15;
      const unsigned short* hp = (const unsigned short*)&rv[rep];
#pragma unroll
      for (int j = 0; j < 8; ++j)
        ldsVT[buf][(c * 8 + j) * 32 + r] = hp[j];
    }
  };

  const float scale = 0.08838834764831845f;  // 1/sqrt(128)
  const int nChunks = (qblk + 64) >> 5;

  asyncLoadK(0, 0);
  loadV(0);
  storeV(0);
  wait_async();
  __syncthreads();

  for (int ch = 0; ch < nChunks; ++ch) {
    const int kv0 = ch * 32;
    const int cur = ch & 1, nxt = cur ^ 1;
    if (ch + 1 < nChunks) {
      asyncLoadK(kv0 + 32, nxt);
      loadV(kv0 + 32);
    }
    if (ch + 2 < nChunks) {
      int r = t >> 2, c = t & 3;
      __builtin_prefetch(Kt + base + (size_t)(kv0 + 64 + r) * rowStride + headOff + c * 32, 0, 3);
      __builtin_prefetch(Vt + base + (size_t)(kv0 + 64 + r) * rowStride + headOff + c * 32, 0, 3);
    }

    // scores = Q (16x128) . K^T (128x32) -> two 16x16 tiles
    v8f s[2];
#pragma unroll
    for (int nt = 0; nt < 2; ++nt) {
      v8f a;
#pragma unroll
      for (int e = 0; e < 8; ++e) a[e] = 0.0f;
      int n = nt * 16 + l16;
#pragma unroll
      for (int c = 0; c < 4; ++c) {
        FragU kfr;
        kfr.q[0] = *(const u32x4*)&ldsK[cur][n * 128 + c * 32 + b0];
        kfr.q[1] = *(const u32x4*)&ldsK[cur][n * 128 + c * 32 + 16 + b0];
        a = __builtin_amdgcn_wmma_f32_16x16x32_bf16(false, qf[c].v, false, kfr.v,
                                                    (short)0, a, false, false);
      }
      s[nt] = a;
    }
    // scale + causal mask
#pragma unroll
    for (int nt = 0; nt < 2; ++nt) {
      int kg = kv0 + nt * 16 + l16;
#pragma unroll
      for (int e = 0; e < 8; ++e) {
        int rg = qblk + wave * 16 + e + hl * 8;
        float v = s[nt][e] * scale;
        s[nt][e] = (kg > rg) ? -1e30f : v;
      }
    }
    // online softmax per owned row; row values live across the 16-lane half
#pragma unroll
    for (int e = 0; e < 8; ++e) {
      float cm = fmaxf(s[0][e], s[1][e]);
#pragma unroll
      for (int off = 1; off <= 8; off <<= 1)
        cm = fmaxf(cm, __shfl_xor(cm, off, 32));
      float mnew  = fmaxf(mrow[e], cm);
      float alpha = __expf(mrow[e] - mnew);
      float p0 = __expf(s[0][e] - mnew);
      float p1 = __expf(s[1][e] - mnew);
      float rs = p0 + p1;
#pragma unroll
      for (int off = 1; off <= 8; off <<= 1)
        rs += __shfl_xor(rs, off, 32);
      lrow[e] = lrow[e] * alpha + rs;
      mrow[e] = mnew;
#pragma unroll
      for (int dt = 0; dt < 8; ++dt) O[dt][e] *= alpha;
      int row = e + hl * 8;
      ldsP[wave * 512 + row * 32 + l16]      = f2bf(p0);
      ldsP[wave * 512 + row * 32 + 16 + l16] = f2bf(p1);
    }
    // P (16x32) as A-fragment (same-wave LDS, DS ops are in-order per wave)
    FragU pf;
    pf.q[0] = *(const u32x4*)&ldsP[wave * 512 + l16 * 32 + b0];
    pf.q[1] = *(const u32x4*)&ldsP[wave * 512 + l16 * 32 + 16 + b0];
    // O += P . V  (V^T staged -> contiguous B-fragments)
#pragma unroll
    for (int dt = 0; dt < 8; ++dt) {
      FragU vfr;
      int d = dt * 16 + l16;
      vfr.q[0] = *(const u32x4*)&ldsVT[cur][d * 32 + b0];
      vfr.q[1] = *(const u32x4*)&ldsVT[cur][d * 32 + 16 + b0];
      O[dt] = __builtin_amdgcn_wmma_f32_16x16x32_bf16(false, pf.v, false, vfr.v,
                                                      (short)0, O[dt], false, false);
    }
    if (ch + 1 < nChunks) storeV(nxt);
    wait_async();
    __syncthreads();
  }

  // normalize and write context (bf16) for the output projection
#pragma unroll
  for (int e = 0; e < 8; ++e) {
    float inv = 1.0f / lrow[e];
    int row = qblk + wave * 16 + e + hl * 8;
    unsigned short* crow = Ctx + base + (size_t)row * rowStride + headOff;
#pragma unroll
    for (int dt = 0; dt < 8; ++dt)
      crow[dt * 16 + l16] = f2bf(O[dt][e] * inv);
  }
}

// ---------------------------------------------------------------------------
// Host orchestration
// ---------------------------------------------------------------------------
extern "C" void kernel_launch(void* const* d_in, const int* in_sizes, int n_in,
                              void* d_out, int out_size, void* d_ws, size_t ws_size,
                              hipStream_t stream) {
  (void)in_sizes; (void)n_in; (void)out_size; (void)ws_size;
  const float* hs   = (const float*)d_in[0];
  const float* wqa  = (const float*)d_in[1];
  const float* wqb  = (const float*)d_in[2];
  const float* wkva = (const float*)d_in[3];
  const float* wkb  = (const float*)d_in[4];
  const float* wvb  = (const float*)d_in[5];
  const float* wo   = (const float*)d_in[6];

  float* outAttn = (float*)d_out;                       // [B*S, H]
  float* outCkv  = (float*)d_out + (size_t)MS * HH;     // [B*S, KVR]

  unsigned short* w = (unsigned short*)d_ws;
  size_t off = 0;
  auto alloc = [&](size_t n) { unsigned short* p = w + off; off += n; return p; };
  unsigned short* hsb   = alloc((size_t)MS * HH);
  unsigned short* wqab  = alloc((size_t)HH * QR);
  unsigned short* wqbb  = alloc((size_t)QR * DM);
  unsigned short* wkvab = alloc((size_t)HH * KVR);
  unsigned short* wkbb  = alloc((size_t)KVR * DM);
  unsigned short* wvbb  = alloc((size_t)KVR * DM);
  unsigned short* wob   = alloc((size_t)DM * HH);
  unsigned short* qcb   = alloc((size_t)MS * QR);
  unsigned short* ckvb  = alloc((size_t)MS * KVR);
  unsigned short* qb    = alloc((size_t)MS * DM);
  unsigned short* kb    = alloc((size_t)MS * DM);
  unsigned short* vb    = alloc((size_t)MS * DM);
  unsigned short* ctxb  = alloc((size_t)MS * DM);

  auto cvt = [&](const float* src, unsigned short* dst, size_t n) {
    int n4 = (int)(n >> 2);
    cvt_f32_bf16<<<(n4 + 255) / 256, 256, 0, stream>>>(src, dst, n4);
  };
  cvt(hs,   hsb,   (size_t)MS * HH);
  cvt(wqa,  wqab,  (size_t)HH * QR);
  cvt(wqb,  wqbb,  (size_t)QR * DM);
  cvt(wkva, wkvab, (size_t)HH * KVR);
  cvt(wkb,  wkbb,  (size_t)KVR * DM);
  cvt(wvb,  wvbb,  (size_t)KVR * DM);
  cvt(wo,   wob,   (size_t)DM * HH);

  dim3 blk(256);
  // q_c = hidden @ w_qa                  [4096 x 1536], K=4096
  gemm_bf16<0><<<dim3(QR / 128, MS / 128), blk, 0, stream>>>(hsb, wqab, qcb, nullptr, MS, QR, HH);
  // compressed_kv = hidden @ w_kva       [4096 x 512],  K=4096  (fp32 out + bf16 feed)
  gemm_bf16<2><<<dim3(KVR / 128, MS / 128), blk, 0, stream>>>(hsb, wkvab, ckvb, outCkv, MS, KVR, HH);
  // q = q_c @ w_qb                       [4096 x 4096], K=1536
  gemm_bf16<0><<<dim3(DM / 128, MS / 128), blk, 0, stream>>>(qcb, wqbb, qb, nullptr, MS, DM, QR);
  // k = ckv @ w_kb                       [4096 x 4096], K=512
  gemm_bf16<0><<<dim3(DM / 128, MS / 128), blk, 0, stream>>>(ckvb, wkbb, kb, nullptr, MS, DM, KVR);
  // v = ckv @ w_vb                       [4096 x 4096], K=512
  gemm_bf16<0><<<dim3(DM / 128, MS / 128), blk, 0, stream>>>(ckvb, wvbb, vb, nullptr, MS, DM, KVR);
  // causal flash attention per (b, h)
  mla_attn<<<dim3(SS / 64, NHH, BB), dim3(128), 0, stream>>>(qb, kb, vb, ctxb);
  // attn_output = ctx @ w_o              [4096 x 4096], K=4096, fp32 out
  gemm_bf16<1><<<dim3(HH / 128, MS / 128), blk, 0, stream>>>(ctxb, wob, nullptr, outAttn, MS, HH, DM);
}
